// RVQEncoder_30640296689693
// MI455X (gfx1250) — matrix operand order repaired and checked by reading, not compiled, not run on hardware
//
#include <hip/hip_runtime.h>
#include <hip/hip_bf16.h>

typedef __attribute__((ext_vector_type(16))) _Float16 v16h;
typedef __attribute__((ext_vector_type(8)))  float    v8f;

#define INPUT_DIM 512
#define CB_SIZE   1024
#define NUM_CB    32
#define DDIM      128
#define NUM_SEM   10
#define SEM_DIM   4096
#define BATCH     8
#define SEQ       4096
#define M_TOTAL   (BATCH*SEQ)   // 32768

// ---------------------------------------------------------------------------
// A-fragment builder: 16x32 f16 A tile per wave, contiguous k order, same as the B packing.
// lane<16: row M=lane,   halves e0..7 -> K=kb+0..7,  e8..15 -> K=kb+8..15
// lane>=16: row M=lane-16, halves e0..7 -> K=kb+16..23, e8..15 -> K=kb+24..31
// ---------------------------------------------------------------------------
__device__ __forceinline__ v16h build_a_frag(const float* __restrict__ rowp,
                                             int kbase, int lane) {
    v16h a;
    int kb0 = kbase + ((lane < 16) ? 0 : 16);
    int kb1 = kbase + ((lane < 16) ? 8 : 24);
#pragma unroll
    for (int e = 0; e < 8; ++e) a[e] = (_Float16)rowp[kb0 + e];
#pragma unroll
    for (int e = 0; e < 8; ++e) a[8 + e] = (_Float16)rowp[kb1 + e];
    return a;
}

// ---------------------------------------------------------------------------
// Pack codebooks [32][1024][128] f32 -> f16 B-fragment layout:
// [t][nchunk(64)][kchunk(4)][lane(32)][e(16)]
// B[K][N]: lane -> N=lane&15, K = kchunk*32 + (lane<16?0:16) + e
// ---------------------------------------------------------------------------
__global__ void pack_cb_kernel(const float* __restrict__ cb, _Float16* __restrict__ out) {
    int tid = blockIdx.x * 256 + threadIdx.x;           // 0..4194303
    int e    = tid & 15;
    int lane = (tid >> 4) & 31;
    int kc   = (tid >> 9) & 3;
    int nc   = (tid >> 11) & 63;
    int t    = tid >> 17;
    int n = nc * 16 + (lane & 15);
    int k = kc * 32 + ((lane < 16) ? 0 : 16) + e;
    out[tid] = (_Float16)cb[((size_t)t * CB_SIZE + n) * DDIM + k];
}

// Pack in_w [128][512] -> f16 fragment layout [nt(8)][kc(16)][lane][e]
__global__ void pack_w_kernel(const float* __restrict__ w, _Float16* __restrict__ out) {
    int tid = blockIdx.x * 256 + threadIdx.x;           // 0..65535
    int e    = tid & 15;
    int lane = (tid >> 4) & 31;
    int kc   = (tid >> 9) & 15;
    int nt   = tid >> 13;
    int n = nt * 16 + (lane & 15);
    int k = kc * 32 + ((lane < 16) ? 0 : 16) + e;
    out[tid] = (_Float16)w[(size_t)n * INPUT_DIM + k];
}

// Codebook row norms: cn[t*1024+n] = sum_d cb^2
__global__ void cbnorm_kernel(const float* __restrict__ cb, float* __restrict__ cn) {
    int tid = blockIdx.x * 256 + threadIdx.x;           // 0..32767
    const float* row = cb + (size_t)tid * DDIM;
    float s = 0.f;
#pragma unroll 8
    for (int c = 0; c < DDIM; ++c) s += row[c] * row[c];
    cn[tid] = s;
}

// Semantic biases: biases[t][b][d]; t>=10 -> 0
__global__ void sem_kernel(const float* __restrict__ sem_w, const float* __restrict__ sem_b,
                           const float* __restrict__ ctx, float* __restrict__ biases) {
    int tid = blockIdx.x * 256 + threadIdx.x;           // 0..32767 = t*1024 + b*128 + d
    int d = tid & 127;
    int b = (tid >> 7) & 7;
    int t = tid >> 10;
    float v = 0.f;
    if (t < NUM_SEM) {
        const float* wr = sem_w + ((size_t)t * DDIM + d) * SEM_DIM;
        const float* cr = ctx + (size_t)b * SEM_DIM;
        float s = 0.f;
#pragma unroll 8
        for (int c = 0; c < SEM_DIM; ++c) s += wr[c] * cr[c];
        v = s + sem_b[t * DDIM + d];
    }
    biases[((size_t)t * BATCH + b) * DDIM + d] = v;
}

// sumbias01[b][d] = 0.1 * sum_t biases[t][b][d]
__global__ void sumbias_kernel(const float* __restrict__ biases, float* __restrict__ sb) {
    int tid = blockIdx.x * 256 + threadIdx.x;           // 0..1023
    float s = 0.f;
#pragma unroll
    for (int t = 0; t < NUM_SEM; ++t) s += biases[(size_t)t * BATCH * DDIM + tid];
    sb[tid] = 0.1f * s;
}

// ---------------------------------------------------------------------------
// x = audio @ in_w^T + in_b  (M=32768, N=128, K=512) via WMMA f16/f32-acc.
// 8 waves/block, each wave: 16 rows x all 128 cols, K loop of 16 chunks.
// Writes x and initializes residual = x.
// ---------------------------------------------------------------------------
__global__ __launch_bounds__(256) void xgemm_kernel(
    const float* __restrict__ audio, const _Float16* __restrict__ wpack,
    const float* __restrict__ in_b, float* __restrict__ x, float* __restrict__ residual) {
    __shared__ float stage[8][512];
    int tid = threadIdx.x;
    int wv = tid >> 5, lane = tid & 31;
    int blockRow0 = blockIdx.x * 128;
    int waveRow0 = blockRow0 + wv * 16;

    v8f acc[8];
#pragma unroll
    for (int nt = 0; nt < 8; ++nt) acc[nt] = (v8f){0.f,0.f,0.f,0.f,0.f,0.f,0.f,0.f};

    for (int kc = 0; kc < 16; ++kc) {
        __syncthreads();
#pragma unroll
        for (int j = 0; j < 16; ++j) {
            int eg = j * 256 + tid;              // 4096 elements (8 waves x 16x32)
            int w2 = eg >> 9, e = eg & 511, row = e >> 5, col = e & 31;
            stage[w2][e] = audio[(size_t)(blockRow0 + w2 * 16 + row) * INPUT_DIM + kc * 32 + col];
        }
        __syncthreads();
        const float* rowp = &stage[wv][(lane & 15) * 32];
        v16h a = build_a_frag(rowp, 0, lane);
#pragma unroll
        for (int nt = 0; nt < 8; ++nt) {
            const _Float16* bp = wpack + ((size_t)(nt * 16 + kc) * 512 + lane * 16);
            v16h b = *(const v16h*)bp;
            acc[nt] = __builtin_amdgcn_wmma_f32_16x16x32_f16(
                false, a, false, b, (short)0, acc[nt], false, false);
        }
    }
    int mbase = (lane >= 16) ? 8 : 0;
#pragma unroll
    for (int nt = 0; nt < 8; ++nt) {
        int N = nt * 16 + (lane & 15);
        float bb = in_b[N];
#pragma unroll
        for (int i = 0; i < 8; ++i) {
            int gm = waveRow0 + mbase + i;
            float v = acc[nt][i] + bb;
            x[(size_t)gm * DDIM + N] = v;
            residual[(size_t)gm * DDIM + N] = v;
        }
    }
}

// ---------------------------------------------------------------------------
// One RVQ step, fully fused: bias add -> distances (WMMA) -> argmin ->
// gather -> residual update -> closs partial. 4 waves/block, 64 rows/block.
// ---------------------------------------------------------------------------
__global__ __launch_bounds__(128) void step_kernel(
    float* __restrict__ residual, const float* __restrict__ x,
    const float* __restrict__ biases_t, const _Float16* __restrict__ cbpack_t,
    const float* __restrict__ cn_t, const float* __restrict__ cb32_t,
    int* __restrict__ idx_out_t, float* __restrict__ partial_t) {
    __shared__ float rows[4][2048];     // 32 KB: 4 waves x 16 rows x 128
    __shared__ float rnorm[4][16];
    __shared__ float redD[4][16][16];
    __shared__ int   redI[4][16][16];
    __shared__ int   sidx[4][16];

    int tid = threadIdx.x;
    int wv = tid >> 5, lane = tid & 31;
    int blockRow0 = blockIdx.x * 64;

    // Stage residual rows + 0.1*bias (coalesced across block)
#pragma unroll 4
    for (int j = 0; j < 64; ++j) {
        int eg = j * 128 + tid;                         // 8192 elements
        int w2 = eg >> 11, e = eg & 2047, col = e & 127;
        int gm = blockRow0 + w2 * 16 + (e >> 7);
        int b = gm >> 12;
        rows[w2][e] = residual[(size_t)gm * DDIM + col] + 0.1f * biases_t[b * DDIM + col];
    }
    __syncthreads();
    // Row norms
    if (tid < 64) {
        int w2 = tid >> 4, m = tid & 15;
        const float* rp = &rows[w2][m * 128];
        float s = 0.f;
#pragma unroll 8
        for (int c = 0; c < DDIM; ++c) s += rp[c] * rp[c];
        rnorm[w2][m] = s;
    }
    __syncthreads();

    // A fragments for this wave's 16 rows (K=128 -> 4 chunks)
    const float* rowp = &rows[wv][(lane & 15) * 128];
    v16h afr[4];
#pragma unroll
    for (int kc = 0; kc < 4; ++kc) afr[kc] = build_a_frag(rowp, kc * 32, lane);

    int mbase = (lane >= 16) ? 8 : 0;
    float rn[8];
#pragma unroll
    for (int i = 0; i < 8; ++i) rn[i] = rnorm[wv][mbase + i];

    float bestd[8];
    int   besti[8];
#pragma unroll
    for (int i = 0; i < 8; ++i) { bestd[i] = 3.4e38f; besti[i] = 0; }

    int ncol = lane & 15;
    for (int nc = 0; nc < 64; ++nc) {
        const _Float16* bp0 = cbpack_t + (size_t)nc * 4 * 512 + lane * 16;
        v16h b0 = *(const v16h*)(bp0);
        v16h b1 = *(const v16h*)(bp0 + 512);
        v16h b2 = *(const v16h*)(bp0 + 1024);
        v16h b3 = *(const v16h*)(bp0 + 1536);
        v8f acc = (v8f){0.f,0.f,0.f,0.f,0.f,0.f,0.f,0.f};
        acc = __builtin_amdgcn_wmma_f32_16x16x32_f16(false, afr[0], false, b0, (short)0, acc, false, false);
        acc = __builtin_amdgcn_wmma_f32_16x16x32_f16(false, afr[1], false, b1, (short)0, acc, false, false);
        acc = __builtin_amdgcn_wmma_f32_16x16x32_f16(false, afr[2], false, b2, (short)0, acc, false, false);
        acc = __builtin_amdgcn_wmma_f32_16x16x32_f16(false, afr[3], false, b3, (short)0, acc, false, false);
        int code = nc * 16 + ncol;
        float cnv = cn_t[code];
#pragma unroll
        for (int i = 0; i < 8; ++i) {
            float d2 = rn[i] - 2.0f * acc[i] + cnv;
            if (d2 < bestd[i]) { bestd[i] = d2; besti[i] = code; }
        }
    }
    // Cross-lane argmin reduce via LDS (tie-break: smaller index, matches jnp.argmin)
#pragma unroll
    for (int i = 0; i < 8; ++i) {
        redD[wv][mbase + i][ncol] = bestd[i];
        redI[wv][mbase + i][ncol] = besti[i];
    }
    __syncthreads();
    if (tid < 64) {
        int w2 = tid >> 4, m = tid & 15;
        float bd = redD[w2][m][0];
        int   bi = redI[w2][m][0];
#pragma unroll
        for (int c = 1; c < 16; ++c) {
            float d2 = redD[w2][m][c];
            int   ii = redI[w2][m][c];
            if (d2 < bd || (d2 == bd && ii < bi)) { bd = d2; bi = ii; }
        }
        sidx[w2][m] = bi;
        idx_out_t[blockRow0 + tid] = bi;    // blockRow0 + w2*16 + m
    }
    __syncthreads();

    // Gather + residual update + closs partial
    float lsum = 0.f;
#pragma unroll 4
    for (int j = 0; j < 64; ++j) {
        int eg = j * 128 + tid;
        int w2 = eg >> 11, e = eg & 2047, col = e & 127;
        int gm = blockRow0 + w2 * 16 + (e >> 7);
        float q = cb32_t[(size_t)sidx[w2][e >> 7] * DDIM + col];
        residual[(size_t)gm * DDIM + col] = rows[w2][e] - q;
        float dq = q - x[(size_t)gm * DDIM + col];
        lsum += dq * dq;
    }
    // Deterministic block reduction (reuse redD as scratch)
    float* red = &redD[0][0][0];
    __syncthreads();
    red[tid] = lsum;
    __syncthreads();
    for (int s = 64; s > 0; s >>= 1) {
        if (tid < s) red[tid] += red[tid + s];
        __syncthreads();
    }
    if (tid == 0) partial_t[blockIdx.x] = red[0];
}

// quantized = sum(qs) = x + 0.1*sum(bias) - residual_final  (STE identity)
__global__ void final_kernel(const float* __restrict__ x, const float* __restrict__ residual,
                             const float* __restrict__ sb, float* __restrict__ qout) {
    size_t tid = (size_t)blockIdx.x * 256 + threadIdx.x;   // 0..4194303
    int d = (int)(tid & 127);
    int b = (int)(tid >> 19);                              // m>>12, m=tid>>7
    qout[tid] = x[tid] + sb[b * DDIM + d] - residual[tid];
}

__global__ void closs_kernel(const float* __restrict__ partials, float* __restrict__ out) {
    if (threadIdx.x == 0 && blockIdx.x == 0) {
        float s = 0.f;
        for (int i = 0; i < NUM_CB * 512; ++i) s += partials[i];
        out[0] = 0.25f * s / ((float)M_TOTAL * (float)DDIM);
    }
}

extern "C" void kernel_launch(void* const* d_in, const int* in_sizes, int n_in,
                              void* d_out, int out_size, void* d_ws, size_t ws_size,
                              hipStream_t stream) {
    const float* audio = (const float*)d_in[0];
    const float* ctx   = (const float*)d_in[1];
    const float* in_w  = (const float*)d_in[2];
    const float* in_b  = (const float*)d_in[3];
    const float* cb    = (const float*)d_in[4];
    const float* sem_w = (const float*)d_in[5];
    const float* sem_b = (const float*)d_in[6];

    char* ws = (char*)d_ws;
    float*    x        = (float*)(ws);                            // 16 MB
    float*    residual = (float*)(ws + (16u << 20));              // 16 MB
    _Float16* cbp      = (_Float16*)(ws + (32u << 20));           // 8 MB
    float*    cn       = (float*)(ws + (40u << 20));              // 128 KB
    _Float16* wpk      = (_Float16*)(ws + (41u << 20));           // 128 KB
    float*    biases   = (float*)(ws + (42u << 20));              // 128 KB
    float*    sb       = (float*)(ws + (43u << 20));              // 4 KB
    float*    partials = (float*)(ws + (43u << 20) + (1u << 18)); // 64 KB

    float* qout      = (float*)d_out;
    int*   idx_out   = (int*)((float*)d_out + (size_t)M_TOTAL * DDIM);
    float* closs_out = (float*)d_out + (size_t)M_TOTAL * DDIM + (size_t)NUM_CB * M_TOTAL;

    pack_cb_kernel<<<16384, 256, 0, stream>>>(cb, cbp);
    cbnorm_kernel<<<128, 256, 0, stream>>>(cb, cn);
    pack_w_kernel<<<256, 256, 0, stream>>>(in_w, wpk);
    sem_kernel<<<128, 256, 0, stream>>>(sem_w, sem_b, ctx, biases);
    sumbias_kernel<<<4, 256, 0, stream>>>(biases, sb);
    xgemm_kernel<<<256, 256, 0, stream>>>(audio, wpk, in_b, x, residual);
    for (int t = 0; t < NUM_CB; ++t) {
        step_kernel<<<512, 128, 0, stream>>>(
            residual, x,
            biases + (size_t)t * BATCH * DDIM,
            cbp + (size_t)t * CB_SIZE * DDIM,
            cn + (size_t)t * CB_SIZE,
            cb + (size_t)t * CB_SIZE * DDIM,
            idx_out + (size_t)t * M_TOTAL,
            partials + (size_t)t * 512);
    }
    final_kernel<<<16384, 256, 0, stream>>>(x, residual, sb, qout);
    closs_kernel<<<1, 64, 0, stream>>>(partials, closs_out);
}
